// ModernBertForRelationExtraction_56264071577855
// MI455X (gfx1250) — matrix operand order repaired and checked by reading, not compiled, not run on hardware
//
#include <hip/hip_runtime.h>

typedef __attribute__((ext_vector_type(16))) __bf16 v16bf;
typedef __attribute__((ext_vector_type(8)))  float  v8f;

#define DEV __device__ __forceinline__

static constexpr int Bn = 4, Sn = 256, Hn = 768, Pn = 54;
static constexpr int BSn = Bn * Sn;      // 1024
static constexpr int H2n = 2 * Hn;       // 1536
static constexpr int KC  = Hn / 32;      // 24 K-chunks
static constexpr float THRESH = 0.5f;

DEV float relu(float v) { return v > 0.f ? v : 0.f; }

DEV v8f wmma_bf16(v16bf a, v16bf b, v8f c) {
  return __builtin_amdgcn_wmma_f32_16x16x32_bf16(
      /*neg_a=*/false, a, /*neg_b=*/false, b,
      /*c_mod=*/(short)0, c, /*reuse_a=*/false, /*reuse_b=*/false);
}

// Pack 8 consecutive fragment elements [BASE..BASE+8) from two float4 runs.
template <int BASE>
DEV void pack8(v16bf& d, float4 u, float4 v) {
  d[BASE + 0] = (__bf16)u.x; d[BASE + 1] = (__bf16)u.y;
  d[BASE + 2] = (__bf16)u.z; d[BASE + 3] = (__bf16)u.w;
  d[BASE + 4] = (__bf16)v.x; d[BASE + 5] = (__bf16)v.y;
  d[BASE + 6] = (__bf16)v.z; d[BASE + 7] = (__bf16)v.w;
}

// Pack 8 elements of relu(u+w) from two float4 runs each.
template <int BASE>
DEV void pack_relu8(v16bf& d, float4 u0, float4 w0, float4 u1, float4 w1) {
  d[BASE + 0] = (__bf16)relu(u0.x + w0.x); d[BASE + 1] = (__bf16)relu(u0.y + w0.y);
  d[BASE + 2] = (__bf16)relu(u0.z + w0.z); d[BASE + 3] = (__bf16)relu(u0.w + w0.w);
  d[BASE + 4] = (__bf16)relu(u1.x + w1.x); d[BASE + 5] = (__bf16)relu(u1.y + w1.y);
  d[BASE + 6] = (__bf16)relu(u1.z + w1.z); d[BASE + 7] = (__bf16)relu(u1.w + w1.w);
}

// Load 16 consecutive fp32 weights and convert into a B fragment.
DEV void load_w16(v16bf& d, const float* __restrict__ p) {
  float4 w0 = *(const float4*)(p);
  float4 w1 = *(const float4*)(p + 4);
  float4 w2 = *(const float4*)(p + 8);
  float4 w3 = *(const float4*)(p + 12);
  pack8<0>(d, w0, w1);
  pack8<8>(d, w2, w3);
}

// -------------------- Kernel 1: candidate gate --------------------
// cand[row] = (relu(x[row]@W1^T + b1) . W2[0] + b2[0]) > THRESH
__global__ void __launch_bounds__(256)
cand_kernel(const float* __restrict__ x, const float* __restrict__ W1s,
            const float* __restrict__ b1s, const float* __restrict__ W2s,
            const float* __restrict__ b2s, int* __restrict__ cand) {
  __shared__ float xr[Hn];
  __shared__ float partial[256];
  const int row = blockIdx.x;                 // 0..1023
  const float* xrow = x + (size_t)row * Hn;
  for (int t = threadIdx.x; t < Hn; t += 256) xr[t] = xrow[t];
  __syncthreads();

  float acc = 0.f;
  for (int k = threadIdx.x; k < Hn; k += 256) {
    const float* wrow = W1s + (size_t)k * Hn;
    float s = b1s[k];
    for (int h = 0; h < Hn; h += 4) {
      float4 w = *(const float4*)(wrow + h);
      float4 xv = *(const float4*)(xr + h);
      s += xv.x * w.x + xv.y * w.y + xv.z * w.z + xv.w * w.w;
    }
    acc += relu(s) * W2s[k];                  // W2s row 0
  }
  partial[threadIdx.x] = acc;
  __syncthreads();
  for (int off = 128; off > 0; off >>= 1) {
    if (threadIdx.x < off) partial[threadIdx.x] += partial[threadIdx.x + off];
    __syncthreads();
  }
  if (threadIdx.x == 0) cand[row] = (partial[0] + b2s[0]) > THRESH ? 1 : 0;
}

// -------------------- Kernel 2: prepack Wp2 into bf16 B-fragments ----------
// Layout: frag[(t*KC + kc)*32 + lane] = 16 bf16 (32 B) for B[k,n] = Wp2[n0+n,k]
// with n = lane&15, k = kc*32 + (lane>=16?16:0) + e (e = 0..15 consecutive).
__global__ void __launch_bounds__(32)
packw_kernel(const float* __restrict__ Wp2, __bf16* __restrict__ W2pack) {
  const int t = blockIdx.x / KC;        // N-tile 0..3
  const int kc = blockIdx.x % KC;       // K-chunk 0..23
  const int lane = threadIdx.x;
  const int half = lane >> 4, l15 = lane & 15;
  const int p = t * 16 + l15;
  const int kbase = kc * 32 + half * 16;

  v16bf f = {};
  if (p < Pn) load_w16(f, Wp2 + (size_t)p * Hn + kbase);
  ((v16bf*)W2pack)[((size_t)t * KC + kc) * 32 + lane] = f;
}

// -------------------- Kernel 3: ha / (hb + bp1) via WMMA --------------------
// One wave = one 16x16 tile of the (1024 x 1536) projection. n0 < 768 -> ha,
// else -> hbb (with bp1 folded in).
__global__ void __launch_bounds__(128)
proj_kernel(const float* __restrict__ x, const float* __restrict__ Wp1,
            const float* __restrict__ bp1, float* __restrict__ ha,
            float* __restrict__ hbb) {
  const int gw = (blockIdx.x * 128 + (int)threadIdx.x) >> 5;   // global wave
  if (gw >= 64 * 96) return;
  const int mt = gw / 96, nt = gw % 96;
  const int m0 = mt * 16, n0 = nt * 16;

  const int lane = threadIdx.x & 31;
  const int half = lane >> 4, l15 = lane & 15;
  const int b0 = half * 16, kb0 = half * 16;

  const bool isA = (n0 < Hn);
  const int c = isA ? (n0 + l15) : (n0 - Hn + l15);   // output channel
  const int koff = isA ? 0 : Hn;

  const float* xrow = x + (size_t)(m0 + l15) * Hn;
  const float* wrow = Wp1 + (size_t)c * H2n + koff;

  v8f acc = {};
#pragma unroll 2
  for (int kk = 0; kk < Hn; kk += 32) {
    v16bf A, Bf;
    {
      const float* p = xrow + kk + b0;
      float4 a0 = *(const float4*)(p);
      float4 a1 = *(const float4*)(p + 4);
      float4 a2 = *(const float4*)(p + 8);
      float4 a3 = *(const float4*)(p + 12);
      pack8<0>(A, a0, a1);
      pack8<8>(A, a2, a3);
    }
    load_w16(Bf, wrow + kk + kb0);
    acc = wmma_bf16(A, Bf, acc);
  }

  if (isA) {
#pragma unroll
    for (int r = 0; r < 8; ++r) {
      int s = m0 + half * 8 + r;
      ha[(size_t)s * Hn + n0 + l15] = acc[r];
    }
  } else {
    float bias = bp1[c];
#pragma unroll
    for (int r = 0; r < 8; ++r) {
      int s = m0 + half * 8 + r;
      hbb[(size_t)s * Hn + c] = acc[r] + bias;
    }
  }
}

// -------------------- Kernel 4: masked pair logits --------------------
// One wave = (b, i, 16 j's) x 64 output channels (P=54 padded to 64).
// K=768 in chunks of 32 -> 24 iters x 4 WMMA. Invalid tiles zero-fill only
// (the dominant path: cand is ~1% dense -> near HBM-write-bound).
__global__ void __launch_bounds__(128)
pair_kernel(const float* __restrict__ ha, const float* __restrict__ hbb,
            const __bf16* __restrict__ W2pack, const float* __restrict__ bp2,
            const int* __restrict__ cand, float* __restrict__ out) {
  const int gw = (blockIdx.x * 128 + (int)threadIdx.x) >> 5;
  if (gw >= Bn * Sn * (Sn / 16)) return;
  const int jt = gw & 15;
  const int i = (gw >> 4) & 255;
  const int b = gw >> 12;
  const int j0 = jt * 16;
  const int lane = threadIdx.x & 31;

  float* otile = out + ((size_t)(b * Sn + i) * Sn + j0) * Pn;  // 16*54 contiguous

  const int ci = cand[b * Sn + i];
  int anyv = 0;
#pragma unroll
  for (int r = 0; r < 16; ++r) {                // wave-uniform scalar check
    int j = j0 + r;
    anyv |= (j != i) ? cand[b * Sn + j] : 0;
  }
  if (!ci || !anyv) {                            // dominant path: wide zero-fill
    const float4 z = make_float4(0.f, 0.f, 0.f, 0.f);
    float4* o4 = (float4*)otile;                 // 16*54/4 = 216 float4
#pragma unroll
    for (int t = 0; t < 6; ++t) o4[lane + 32 * t] = z;
    if (lane < 24) o4[lane + 192] = z;
    return;
  }

  const int half = lane >> 4, l15 = lane & 15;
  const int b0 = half * 16;

  const float* haRow = ha + (size_t)(b * Sn + i) * Hn;
  const float* hbRow = hbb + (size_t)(b * Sn + j0 + l15) * Hn;
  const int p0 = l15, p1 = 16 + l15, p2 = 32 + l15, p3 = 48 + l15;

  const v16bf* Wf = (const v16bf*)W2pack;
  __builtin_prefetch(Wf + lane, 0, 0);           // global_prefetch_b8

  // per-lane row-validity mask for the 8 rows this lane's D registers cover
  int rvmask = 0;
#pragma unroll
  for (int r = 0; r < 8; ++r) {
    int j = j0 + half * 8 + r;
    if ((j != i) && cand[b * Sn + j]) rvmask |= (1 << r);
  }

  v8f acc0 = {}, acc1 = {}, acc2 = {}, acc3 = {};
#pragma unroll 4
  for (int kc = 0; kc < KC; ++kc) {
    const int kk = kc * 32;
    v16bf A;
    {
      const float* pa = haRow + kk + b0;
      const float* pb = hbRow + kk + b0;
      float4 a0 = *(const float4*)(pa);      float4 c0 = *(const float4*)(pb);
      float4 a1 = *(const float4*)(pa + 4);  float4 c1 = *(const float4*)(pb + 4);
      float4 a2 = *(const float4*)(pa + 8);  float4 c2 = *(const float4*)(pb + 8);
      float4 a3 = *(const float4*)(pa + 12); float4 c3 = *(const float4*)(pb + 12);
      pack_relu8<0>(A, a0, c0, a1, c1);
      pack_relu8<8>(A, a2, c2, a3, c3);
    }
    // Prepacked B fragments: one 32-B per-lane load each, no conversion.
    acc0 = wmma_bf16(A, Wf[((size_t)0 * KC + kc) * 32 + lane], acc0);
    acc1 = wmma_bf16(A, Wf[((size_t)1 * KC + kc) * 32 + lane], acc1);
    acc2 = wmma_bf16(A, Wf[((size_t)2 * KC + kc) * 32 + lane], acc2);
    acc3 = wmma_bf16(A, Wf[((size_t)3 * KC + kc) * 32 + lane], acc3);
  }

  // Epilogue: D layout lane l15 = N, rows M = half*8 + r.
  {
    float bias = bp2[p0];
#pragma unroll
    for (int r = 0; r < 8; ++r)
      otile[(size_t)(half * 8 + r) * Pn + p0] =
          ((rvmask >> r) & 1) ? (acc0[r] + bias) : 0.f;
  }
  {
    float bias = bp2[p1];
#pragma unroll
    for (int r = 0; r < 8; ++r)
      otile[(size_t)(half * 8 + r) * Pn + p1] =
          ((rvmask >> r) & 1) ? (acc1[r] + bias) : 0.f;
  }
  {
    float bias = bp2[p2];
#pragma unroll
    for (int r = 0; r < 8; ++r)
      otile[(size_t)(half * 8 + r) * Pn + p2] =
          ((rvmask >> r) & 1) ? (acc2[r] + bias) : 0.f;
  }
  if (p3 < Pn) {
    float bias = bp2[p3];
#pragma unroll
    for (int r = 0; r < 8; ++r)
      otile[(size_t)(half * 8 + r) * Pn + p3] =
          ((rvmask >> r) & 1) ? (acc3[r] + bias) : 0.f;
  }
}

// -------------------- launch --------------------
extern "C" void kernel_launch(void* const* d_in, const int* in_sizes, int n_in,
                              void* d_out, int out_size, void* d_ws, size_t ws_size,
                              hipStream_t stream) {
  const float* x   = (const float*)d_in[0];  // (4,256,768)
  const float* W1s = (const float*)d_in[1];  // (768,768)
  const float* b1s = (const float*)d_in[2];  // (768,)
  const float* W2s = (const float*)d_in[3];  // (2,768)
  const float* b2s = (const float*)d_in[4];  // (2,)
  const float* Wp1 = (const float*)d_in[5];  // (768,1536)
  const float* bp1 = (const float*)d_in[6];  // (768,)
  const float* Wp2 = (const float*)d_in[7];  // (54,768)
  const float* bp2 = (const float*)d_in[8];  // (54,)
  float* out = (float*)d_out;                // (4,256,256,54) fp32

  char* ws = (char*)d_ws;
  int*    cand   = (int*)ws;                                        // 4 KB
  float*  ha     = (float*)(ws + 4096);                             // 3 MB
  float*  hbb    = (float*)(ws + 4096 + (size_t)BSn * Hn * 4);      // 3 MB
  __bf16* W2pack = (__bf16*)(ws + 4096 + (size_t)2 * BSn * Hn * 4); // 96 KB

  cand_kernel<<<BSn, 256, 0, stream>>>(x, W1s, b1s, W2s, b2s, cand);
  packw_kernel<<<4 * KC, 32, 0, stream>>>(Wp2, W2pack);
  // 64 M-tiles * 96 N-tiles = 6144 waves -> 1536 blocks of 4 waves
  proj_kernel<<<1536, 128, 0, stream>>>(x, Wp1, bp1, ha, hbb);
  // 4*256*16 = 16384 waves -> 4096 blocks of 4 waves
  pair_kernel<<<4096, 128, 0, stream>>>(ha, hbb, W2pack, bp2, cand, out);
}
